// NERNet_27315992002668
// MI455X (gfx1250) — compile-verified, matrix operand order checked by reading, not run on hardware
//
#include <hip/hip_runtime.h>

// ---------------- model constants ----------------
#define BSZ    64
#define SEQ    512
#define HID    512
#define G4     2048            // 4*H
#define DIN0   256
#define DIN1   1024
#define NTAG   9
#define BSROWS (BSZ * SEQ)     // 32768 = flattened (B,S)

// ---------------- WMMA types ----------------
typedef __attribute__((ext_vector_type(16))) __bf16 bf16x16;
typedef __attribute__((ext_vector_type(8)))  __bf16 bf16x8;
typedef __attribute__((ext_vector_type(8)))  float  f32x8;

union FragAB { bf16x16 v; bf16x8 h[2]; };

__device__ __forceinline__ unsigned short f2bf(float f) {
  union { float f; unsigned u; } v; v.f = f;
  unsigned r = v.u + 0x7fffu + ((v.u >> 16) & 1u);   // round-to-nearest-even
  return (unsigned short)(r >> 16);
}
__device__ __forceinline__ float sigm(float x) { return 1.0f / (1.0f + __expf(-x)); }

// LDS byte-offset of a __shared__ pointer (generic-pointer low 32 bits == LDS offset)
__device__ __forceinline__ unsigned lds_off(const void* p) {
  return (unsigned)(size_t)p;
}
// CDNA5 async global->LDS 16B copy (per-lane), tracked by ASYNCcnt
__device__ __forceinline__ void async_copy_b128(unsigned ldsdst, const void* gsrc) {
  asm volatile("global_load_async_to_lds_b128 %0, %1, off"
               :: "v"(ldsdst), "v"(gsrc) : "memory");
}
__device__ __forceinline__ void wait_async_le4() {
  asm volatile("s_wait_asynccnt 0x4" ::: "memory");
}
__device__ __forceinline__ void wait_async_le0() {
  asm volatile("s_wait_asynccnt 0x0" ::: "memory");
}

// ---------------- utility kernels ----------------
__global__ void zero_u32(unsigned int* p, int n) {
  int i = blockIdx.x * blockDim.x + threadIdx.x;
  if (i < n) p[i] = 0u;
}

__global__ void cvt_bf16(const float* __restrict__ in, unsigned short* __restrict__ out, int n) {
  int i = blockIdx.x * blockDim.x + threadIdx.x;
  if (i < n) out[i] = f2bf(in[i]);
}

// embedding gather + concat -> bf16 (B*S, 256)
__global__ void gather_emb(const int* __restrict__ cid, const int* __restrict__ bid,
                           const float* __restrict__ ctab, const float* __restrict__ btab,
                           unsigned short* __restrict__ out) {
  int i = blockIdx.x * blockDim.x + threadIdx.x;   // over B*S*128
  int pos = i >> 7;
  int d   = i & 127;
  out[(size_t)pos * 256 + d]       = f2bf(ctab[(size_t)cid[pos] * 128 + d]);
  out[(size_t)pos * 256 + 128 + d] = f2bf(btab[(size_t)bid[pos] * 128 + d]);
}

// ---------------- big projection GEMM: C(M,N) = A(M,K) * W(N,K)^T + bias ----------------
// A, W bf16 row-major; C fp32. M%128==0, N%128==0, K%32==0.
// Double-buffered LDS, filled with CDNA5 async global->LDS DMA (ASYNCcnt).
#define TM  128
#define TN  128
#define TK  32
#define LDP 40   // padded LDS row stride in halves (80B: 16B-aligned, bank-spread)

__global__ __launch_bounds__(256) void gemm_bias(
    const unsigned short* __restrict__ A,
    const unsigned short* __restrict__ W,
    const float* __restrict__ bias,
    float* __restrict__ C,
    int M, int N, int K) {
  __shared__ unsigned short lA[2][TM * LDP];
  __shared__ unsigned short lB[2][TN * LDP];
  const int tid  = threadIdx.x;
  const int lane = tid & 31;
  const int wave = tid >> 5;           // 8 waves
  const int wm   = wave & 3;           // 4 M-slabs of 32 rows
  const int wn   = wave >> 2;          // 2 N-slabs of 64 cols
  const int m0   = blockIdx.y * TM;
  const int n0   = blockIdx.x * TN;
  const int r    = lane & 15;
  const int koff = (lane >> 4) * 8;    // wave32 bf16 fragment K-half select

  // this thread's two 8-half chunks per tile (512 chunks of 16B per 128x32 tile)
  const int row0 = (tid)        >> 2, col0 = ((tid)        & 3) * 8;
  const int row1 = (tid + 256)  >> 2, col1 = ((tid + 256)  & 3) * 8;

  const f32x8 zf = {0.f,0.f,0.f,0.f,0.f,0.f,0.f,0.f};
  f32x8 acc[2][4];
#pragma unroll
  for (int ti = 0; ti < 2; ++ti)
#pragma unroll
    for (int tj = 0; tj < 4; ++tj) acc[ti][tj] = zf;

  // issue tile 0 into buffer 0 (4 async 16B chunks per thread: 2 of A, 2 of B)
  async_copy_b128(lds_off(&lA[0][row0 * LDP + col0]), A + (size_t)(m0 + row0) * K + col0);
  async_copy_b128(lds_off(&lB[0][row0 * LDP + col0]), W + (size_t)(n0 + row0) * K + col0);
  async_copy_b128(lds_off(&lA[0][row1 * LDP + col1]), A + (size_t)(m0 + row1) * K + col1);
  async_copy_b128(lds_off(&lB[0][row1 * LDP + col1]), W + (size_t)(n0 + row1) * K + col1);

  for (int kk = 0; kk < K; kk += TK) {
    const int buf = (kk / TK) & 1;
    __syncthreads();                    // readers of buf^1 (prev iter) are done
    if (kk + TK < K) {                  // overlap next tile's DMA with this tile's WMMA
      int kn = kk + TK;
      async_copy_b128(lds_off(&lA[buf ^ 1][row0 * LDP + col0]),
                      A + (size_t)(m0 + row0) * K + kn + col0);
      async_copy_b128(lds_off(&lB[buf ^ 1][row0 * LDP + col0]),
                      W + (size_t)(n0 + row0) * K + kn + col0);
      async_copy_b128(lds_off(&lA[buf ^ 1][row1 * LDP + col1]),
                      A + (size_t)(m0 + row1) * K + kn + col1);
      async_copy_b128(lds_off(&lB[buf ^ 1][row1 * LDP + col1]),
                      W + (size_t)(n0 + row1) * K + kn + col1);
      wait_async_le4();                 // in-order: own tile-kk chunks have landed
    } else {
      wait_async_le0();
    }
    __syncthreads();                    // all waves' tile-kk chunks have landed

    FragAB af[2], bfr[4];
#pragma unroll
    for (int ti = 0; ti < 2; ++ti) {
      const unsigned short* p = &lA[buf][(wm * 32 + ti * 16 + r) * LDP + koff];
      af[ti].h[0] = *(const bf16x8*)(p);
      af[ti].h[1] = *(const bf16x8*)(p + 16);
    }
#pragma unroll
    for (int tj = 0; tj < 4; ++tj) {
      const unsigned short* p = &lB[buf][(wn * 64 + tj * 16 + r) * LDP + koff];
      bfr[tj].h[0] = *(const bf16x8*)(p);
      bfr[tj].h[1] = *(const bf16x8*)(p + 16);
    }
#pragma unroll
    for (int ti = 0; ti < 2; ++ti)
#pragma unroll
      for (int tj = 0; tj < 4; ++tj)
        acc[ti][tj] = __builtin_amdgcn_wmma_f32_16x16x32_bf16(
            false, af[ti].v, false, bfr[tj].v, (short)0, acc[ti][tj], false, false);
  }

#pragma unroll
  for (int ti = 0; ti < 2; ++ti)
#pragma unroll
    for (int tj = 0; tj < 4; ++tj)
#pragma unroll
      for (int e = 0; e < 8; ++e) {
        int m = m0 + wm * 32 + ti * 16 + (lane >> 4) * 8 + e;
        int n = n0 + wn * 64 + tj * 16 + (lane & 15);
        C[(size_t)m * N + n] = acc[ti][tj][e] + bias[n];
      }
}

// ---------------- fused LSTM step ----------------
// One block owns 16 hidden columns x all 4 gates x all 64 batch rows.
// g = xp[:,t] + h_in * whh^T ; gates ; update c,h ; write h (bf16 state + sequence out).
__global__ __launch_bounds__(256) void lstm_step(
    const float* __restrict__ xp,              // (B,S,4H) fp32
    const unsigned short* __restrict__ whh,    // (4H,H) bf16
    const unsigned short* __restrict__ h_in,   // (B,H) bf16 (read)
    unsigned short* __restrict__ h_out,        // (B,H) bf16 (write, ping-pong)
    float* __restrict__ c_state,               // (B,H) fp32
    unsigned short* __restrict__ seq_bf,       // layer0 out (B,S,2H) bf16, or null
    float* __restrict__ seq_f32,               // layer1 out (B,S,2H) fp32, or null
    int tmem, int dircol) {
  __shared__ float gl[64][64];                 // [batch][gate*16 + nlocal]
  const int tid  = threadIdx.x;
  const int lane = tid & 31;
  const int wave = tid >> 5;
  const int wm   = wave & 3;                   // 4 batch slabs of 16
  const int wgrp = wave >> 2;                  // gate pair {0,1} or {2,3}
  const int n0   = blockIdx.x * 16;
  const int r    = lane & 15;
  const int koff = (lane >> 4) * 8;

  const f32x8 zf = {0.f,0.f,0.f,0.f,0.f,0.f,0.f,0.f};
  f32x8 acc[2]; acc[0] = zf; acc[1] = zf;

  for (int kk = 0; kk < HID; kk += 32) {
    FragAB a;
    const unsigned short* pa = h_in + (size_t)(wm * 16 + r) * HID + kk + koff;
    a.h[0] = *(const bf16x8*)(pa);
    a.h[1] = *(const bf16x8*)(pa + 16);
#pragma unroll
    for (int gi = 0; gi < 2; ++gi) {
      int g = wgrp * 2 + gi;
      const unsigned short* pb = whh + (size_t)(g * HID + n0 + r) * HID + kk + koff;
      FragAB b;
      b.h[0] = *(const bf16x8*)(pb);
      b.h[1] = *(const bf16x8*)(pb + 16);
      acc[gi] = __builtin_amdgcn_wmma_f32_16x16x32_bf16(
          false, a.v, false, b.v, (short)0, acc[gi], false, false);
    }
  }

#pragma unroll
  for (int gi = 0; gi < 2; ++gi) {
    int g = wgrp * 2 + gi;
#pragma unroll
    for (int e = 0; e < 8; ++e) {
      int m  = wm * 16 + (lane >> 4) * 8 + e;
      int nl = lane & 15;
      int col = g * HID + n0 + nl;
      gl[m][g * 16 + nl] = acc[gi][e] + xp[((size_t)m * SEQ + tmem) * G4 + col];
    }
  }
  __syncthreads();

#pragma unroll
  for (int q = 0; q < 4; ++q) {
    int idx = tid + q * 256;                   // 1024 (b, j) cells
    int m = idx >> 4;
    int j = idx & 15;
    float iv = sigm(gl[m][j]);
    float fv = sigm(gl[m][16 + j]);
    float gv = tanhf(gl[m][32 + j]);
    float ov = sigm(gl[m][48 + j]);
    int hix = m * HID + n0 + j;
    float c = fv * c_state[hix] + iv * gv;
    c_state[hix] = c;
    float h = ov * tanhf(c);
    h_out[hix] = f2bf(h);
    size_t so = ((size_t)m * SEQ + tmem) * 1024 + dircol + n0 + j;
    if (seq_bf) seq_bf[so] = f2bf(h);
    else        seq_f32[so] = h;
  }
}

// ---------------- emissions: (B*S,1024) x (9,1024)^T + b ----------------
__global__ __launch_bounds__(256) void emission_k(
    const float* __restrict__ h1, const float* __restrict__ ew,
    const float* __restrict__ ebias, float* __restrict__ em) {
  int pos  = blockIdx.x * 8 + (threadIdx.x >> 5);   // (b*S+s), one wave each
  int lane = threadIdx.x & 31;
  const float* hp = h1 + (size_t)pos * 1024;
  float acc[NTAG];
#pragma unroll
  for (int t = 0; t < NTAG; ++t) acc[t] = 0.f;
  for (int k = lane; k < 1024; k += 32) {
    float hv = hp[k];
#pragma unroll
    for (int t = 0; t < NTAG; ++t) acc[t] += hv * ew[t * 1024 + k];
  }
#pragma unroll
  for (int t = 0; t < NTAG; ++t)
#pragma unroll
    for (int o = 16; o > 0; o >>= 1) acc[t] += __shfl_xor(acc[t], o, 32);
  if (lane == 0) {
#pragma unroll
    for (int t = 0; t < NTAG; ++t) em[(size_t)pos * NTAG + t] = acc[t] + ebias[t];
  }
}

// ---------------- CRF ----------------
__global__ void crf_num_k(const float* __restrict__ em, const int* __restrict__ tags,
                          const float* __restrict__ st, const float* __restrict__ en,
                          const float* __restrict__ tr, float* __restrict__ numbuf) {
  int b = threadIdx.x;
  if (b >= BSZ) return;
  const int* tg = tags + b * SEQ;
  const float* e = em + (size_t)b * SEQ * NTAG;
  int prev = tg[0];
  float num = st[prev] + e[prev];
  for (int s = 1; s < SEQ; ++s) {
    int cur = tg[s];
    num += tr[prev * NTAG + cur] + e[s * NTAG + cur];
    prev = cur;
  }
  num += en[prev];
  numbuf[b] = num;
}

__global__ void crf_fwd_k(const float* __restrict__ em, const float* __restrict__ st,
                          const float* __restrict__ en, const float* __restrict__ tr,
                          float* __restrict__ denbuf) {
  int b = blockIdx.x;              // one wave32 per batch; lanes 0..8 carry states
  int j = threadIdx.x;
  int jj = (j < NTAG) ? j : 0;
  const float* e = em + (size_t)b * SEQ * NTAG;
  float score = (j < NTAG) ? (st[jj] + e[jj]) : -1.0e30f;
  for (int t = 1; t < SEQ; ++t) {
    float v[NTAG];
    float mx = -1.0e30f;
#pragma unroll
    for (int i = 0; i < NTAG; ++i) {
      v[i] = __shfl(score, i, 32) + tr[i * NTAG + jj];
      mx = fmaxf(mx, v[i]);
    }
    float sm = 0.f;
#pragma unroll
    for (int i = 0; i < NTAG; ++i) sm += __expf(v[i] - mx);
    float nxt = mx + __logf(sm) + e[t * NTAG + jj];
    score = (j < NTAG) ? nxt : -1.0e30f;
  }
  float z = (j < NTAG) ? (score + en[jj]) : -1.0e30f;
  float mx = z;
#pragma unroll
  for (int o = 16; o > 0; o >>= 1) mx = fmaxf(mx, __shfl_xor(mx, o, 32));
  float ex = (j < NTAG) ? __expf(z - mx) : 0.f;
#pragma unroll
  for (int o = 16; o > 0; o >>= 1) ex += __shfl_xor(ex, o, 32);
  if (j == 0) denbuf[b] = mx + __logf(ex);
}

__global__ void crf_final_k(const float* __restrict__ numbuf, const float* __restrict__ denbuf,
                            float* __restrict__ out) {
  __shared__ float s[BSZ];
  int b = threadIdx.x;
  s[b] = numbuf[b] - denbuf[b];
  __syncthreads();
  if (b == 0) {
    float acc = 0.f;
    for (int i = 0; i < BSZ; ++i) acc += s[i];
    out[0] = -acc / (float)BSZ;
  }
}

// ---------------- host orchestration ----------------
extern "C" void kernel_launch(void* const* d_in, const int* in_sizes, int n_in,
                              void* d_out, int out_size, void* d_ws, size_t ws_size,
                              hipStream_t stream) {
  (void)in_sizes; (void)n_in; (void)out_size; (void)ws_size;

  const int*   char_id   = (const int*)d_in[0];
  const int*   bichar_id = (const int*)d_in[1];
  const int*   labels    = (const int*)d_in[2];
  const float* ctab      = (const float*)d_in[3];
  const float* btab      = (const float*)d_in[4];
  const float* wih[2][2] = {{(const float*)d_in[5],  (const float*)d_in[8]},
                            {(const float*)d_in[11], (const float*)d_in[14]}};
  const float* whh[2][2] = {{(const float*)d_in[6],  (const float*)d_in[9]},
                            {(const float*)d_in[12], (const float*)d_in[15]}};
  const float* bvec[2][2]= {{(const float*)d_in[7],  (const float*)d_in[10]},
                            {(const float*)d_in[13], (const float*)d_in[16]}};
  const float* emit_w = (const float*)d_in[17];
  const float* emit_b = (const float*)d_in[18];
  const float* crf_s  = (const float*)d_in[19];
  const float* crf_e  = (const float*)d_in[20];
  const float* crf_t  = (const float*)d_in[21];

  char* ws = (char*)d_ws;
  size_t off = 0;
  auto alloc = [&](size_t b) -> void* {
    void* p = ws + off;
    off += (b + 255) & ~(size_t)255;
    return p;
  };

  const int wihK[2] = {DIN0, DIN1};
  unsigned short* emb_bf = (unsigned short*)alloc((size_t)BSROWS * 256 * 2);
  unsigned short* wih_bf[2][2];
  unsigned short* whh_bf[2][2];
  for (int l = 0; l < 2; ++l)
    for (int d = 0; d < 2; ++d) {
      wih_bf[l][d] = (unsigned short*)alloc((size_t)G4 * wihK[l] * 2);
      whh_bf[l][d] = (unsigned short*)alloc((size_t)G4 * HID * 2);
    }
  float* xp[2];
  xp[0] = (float*)alloc((size_t)BSROWS * G4 * 4);
  xp[1] = (float*)alloc((size_t)BSROWS * G4 * 4);
  unsigned short* h0_bf = (unsigned short*)alloc((size_t)BSROWS * 1024 * 2);
  float* h1 = (float*)alloc((size_t)BSROWS * 1024 * 4);

  char* stblk = (char*)alloc(524288);   // 4 x h-state(bf16) + 2 x c-state(fp32)
  unsigned short* hstate[2][2];
  hstate[0][0] = (unsigned short*)(stblk);
  hstate[0][1] = (unsigned short*)(stblk + 65536);
  hstate[1][0] = (unsigned short*)(stblk + 131072);
  hstate[1][1] = (unsigned short*)(stblk + 196608);
  float* cstate[2];
  cstate[0] = (float*)(stblk + 262144);
  cstate[1] = (float*)(stblk + 393216);

  float* emis   = (float*)alloc((size_t)BSROWS * NTAG * 4);
  float* numbuf = (float*)alloc(BSZ * 4);
  float* denbuf = (float*)alloc(BSZ * 4);

  // 1) embeddings + weight conversion
  gather_emb<<<BSROWS * 128 / 256, 256, 0, stream>>>(char_id, bichar_id, ctab, btab, emb_bf);
  for (int l = 0; l < 2; ++l)
    for (int d = 0; d < 2; ++d) {
      int n1 = G4 * wihK[l];
      cvt_bf16<<<(n1 + 255) / 256, 256, 0, stream>>>(wih[l][d], wih_bf[l][d], n1);
      int n2 = G4 * HID;
      cvt_bf16<<<(n2 + 255) / 256, 256, 0, stream>>>(whh[l][d], whh_bf[l][d], n2);
    }

  // 2) two BiLSTM layers
  for (int l = 0; l < 2; ++l) {
    const unsigned short* Ain = (l == 0) ? emb_bf : h0_bf;
    int K = wihK[l];
    dim3 grid(G4 / TN, BSROWS / TM);
    gemm_bias<<<grid, 256, 0, stream>>>(Ain, wih_bf[l][0], bvec[l][0], xp[0], BSROWS, G4, K);
    gemm_bias<<<grid, 256, 0, stream>>>(Ain, wih_bf[l][1], bvec[l][1], xp[1], BSROWS, G4, K);

    zero_u32<<<(131072 + 255) / 256, 256, 0, stream>>>((unsigned int*)stblk, 131072);

    unsigned short* sbf = (l == 0) ? h0_bf : (unsigned short*)nullptr;
    float*          sf  = (l == 0) ? (float*)nullptr : h1;
    for (int t = 0; t < SEQ; ++t) {
      int pr = t & 1, nx = (t + 1) & 1;
      lstm_step<<<HID / 16, 256, 0, stream>>>(xp[0], whh_bf[l][0], hstate[0][pr], hstate[0][nx],
                                              cstate[0], sbf, sf, t, 0);
      lstm_step<<<HID / 16, 256, 0, stream>>>(xp[1], whh_bf[l][1], hstate[1][pr], hstate[1][nx],
                                              cstate[1], sbf, sf, SEQ - 1 - t, HID);
    }
  }

  // 3) emissions + CRF NLL
  emission_k<<<BSROWS / 8, 256, 0, stream>>>(h1, emit_w, emit_b, emis);
  crf_num_k<<<1, 64, 0, stream>>>(emis, labels, crf_s, crf_e, crf_t, numbuf);
  crf_fwd_k<<<64, 32, 0, stream>>>(emis, crf_s, crf_e, crf_t, denbuf);
  crf_final_k<<<1, 64, 0, stream>>>(numbuf, denbuf, (float*)d_out);
}
